// MessagePassing_55559696941642
// MI455X (gfx1250) — matrix operand order repaired and checked by reading, not compiled, not run on hardware
//
#include <hip/hip_runtime.h>
#include <hip/hip_bf16.h>

// GIN layer for MI455X (gfx1250, wave32, WMMA).
// f16 WMMA w/ f32 accumulation for adj@x (537MB adj stream @23.3TB/s ~ 23us
// floor), double-buffered LDS staging of adj, 64-row tiles to halve L2 traffic
// on the reused xT operand; fused eps-residual; fused WMMA 2-layer MLP.

typedef _Float16 f16;
typedef __attribute__((ext_vector_type(16))) _Float16 v16h;
typedef __attribute__((ext_vector_type(8)))  _Float16 v8h;
typedef __attribute__((ext_vector_type(8)))  float    v8f;

// ---------------------------------------------------------------------------
// WMMA fragment helpers (wave32; A/B use this code's contiguous k order (same in both), C/D per CDNA5 ISA 7.12.2)
// A (16x32 f16): lane m=l&15; lanes<16 hold K 0..15,
//                lanes>=16 hold K 16..31  -> two aligned 16B loads.
// B (32x16 f16): lane n=l&15; lanes<16 hold K 0..15, lanes>=16 K 16..31
//                -> one contiguous 32B run (two 16B loads).
// C/D (16x16 f32): VGPR r -> row r + (l>=16 ? 8 : 0), col = l&15.
// ---------------------------------------------------------------------------
__device__ __forceinline__ v16h load_a_frag(const _Float16* rowPtr, int hk) {
  const v8h lo = *(const v8h*)(rowPtr + 2 * hk);
  const v8h hi = *(const v8h*)(rowPtr + 2 * hk + 8);
  return __builtin_shufflevector(lo, hi, 0,1,2,3,4,5,6,7,8,9,10,11,12,13,14,15);
}

__device__ __forceinline__ v16h load_b_frag(const _Float16* p) {
  const v8h lo = *(const v8h*)(p);
  const v8h hi = *(const v8h*)(p + 8);
  return __builtin_shufflevector(lo, hi, 0,1,2,3,4,5,6,7,8,9,10,11,12,13,14,15);
}

__device__ __forceinline__ v8f wmma_f16(v16h a, v16h b, v8f c) {
  return __builtin_amdgcn_wmma_f32_16x16x32_f16(false, a, false, b,
                                                (short)0, c, false, false);
}

// ---------------------------------------------------------------------------
// Prep: x [8][4096][256] f32 -> xT [8][256][4096] f16 (tiled transpose)
// ---------------------------------------------------------------------------
__global__ __launch_bounds__(256)
void transpose_x_kernel(const float* __restrict__ x, f16* __restrict__ xT) {
  const int b  = blockIdx.z;
  const int m0 = blockIdx.x * 32;
  const int d0 = blockIdx.y * 32;
  __shared__ f16 tile[32][33];
  #pragma unroll
  for (int j = 0; j < 4; ++j) {
    int m = m0 + threadIdx.y + j * 8;
    tile[threadIdx.y + j * 8][threadIdx.x] =
        (f16)x[((size_t)b * 4096 + m) * 256 + d0 + threadIdx.x];
  }
  __syncthreads();
  #pragma unroll
  for (int j = 0; j < 4; ++j) {
    int d = d0 + threadIdx.y + j * 8;
    xT[((size_t)b * 256 + d) * 4096 + m0 + threadIdx.x] =
        tile[threadIdx.x][threadIdx.y + j * 8];
  }
}

// Prep: W [K][N] f32 -> Wt [N][K] f16
__global__ __launch_bounds__(256)
void transpose_w_kernel(const float* __restrict__ w, f16* __restrict__ wt,
                        int K, int N) {
  int idx = blockIdx.x * 256 + threadIdx.x;
  if (idx < K * N) {
    int n = idx / K;
    int k = idx - n * K;
    wt[idx] = (f16)w[(size_t)k * N + n];
  }
}

// ---------------------------------------------------------------------------
// Stage 1: h = (1+eps)*x + adj @ x  (per batch, h stored f16)
// WG = 256 thr (8 waves), tile = 64 rows x 256 cols, K staged 64 per barrier,
// ping-pong LDS so next adj tile's global loads overlap current WMMAs.
// NOTE: LDS store target is indexed through the __shared__ array directly so
// the compiler keeps the LDS address space (ds_store, not flat_store).
// ---------------------------------------------------------------------------
#define SA2 72   // adj LDS row stride in halves (64 + 8 pad, 16B-aligned rows)

__global__ __launch_bounds__(256)
void gin_bmm_kernel(const float* __restrict__ adj, const f16* __restrict__ xT,
                    const float* __restrict__ x, const float* __restrict__ epsp,
                    f16* __restrict__ hh) {
  const int b    = blockIdx.y;
  const int n0   = blockIdx.x * 64;
  const int tid  = threadIdx.x;
  const int wave = tid >> 5;
  const int lane = tid & 31;
  const int m16  = lane & 15;
  const int hk   = (lane >> 4) * 8;    // A-frag K sub-base
  const int kh16 = (lane >> 4) * 16;   // B-frag K sub-base

  const float* adjB = adj + ((size_t)b * 4096 + n0) * 4096;
  const f16*   xTB  = xT + (size_t)b * 256 * 4096;
  const float* xB   = x  + (size_t)b * 4096 * 256;
  f16*         hhB  = hh + (size_t)b * 4096 * 256;

  __shared__ _Float16 ldsA[2][64 * SA2];

  const v8f vz = {};
  v8f acc[8] = {vz, vz, vz, vz, vz, vz, vz, vz};  // [rb 0..3][cb 0..1]

  // cooperative adj staging: thread loads 64B (16 floats) of one row
  const int lr = tid >> 2;          // tile row 0..63
  const int lc = (tid & 3) * 16;    // 16-float column group
  const int stOff = lr * SA2 + lc;  // LDS offset within a buffer
  const float* adjRow = adjB + (size_t)lr * 4096 + lc;

  {  // prologue: stage chunk 0 into buffer 0
    const float4* s = (const float4*)adjRow;
    float4 a = s[0], bb = s[1], c = s[2], e = s[3];
    _Float16* d = &ldsA[0][stOff];
    d[0]  = (f16)a.x;  d[1]  = (f16)a.y;  d[2]  = (f16)a.z;  d[3]  = (f16)a.w;
    d[4]  = (f16)bb.x; d[5]  = (f16)bb.y; d[6]  = (f16)bb.z; d[7]  = (f16)bb.w;
    d[8]  = (f16)c.x;  d[9]  = (f16)c.y;  d[10] = (f16)c.z;  d[11] = (f16)c.w;
    d[12] = (f16)e.x;  d[13] = (f16)e.y;  d[14] = (f16)e.z;  d[15] = (f16)e.w;
  }
  __syncthreads();

  for (int k0 = 0; k0 < 4096; k0 += 64) {
    const int  p    = (k0 >> 6) & 1;
    const bool more = (k0 + 64) < 4096;

    // issue next tile's global loads first (overlap with WMMAs below)
    float4 g0, g1, g2, g3;
    if (more) {
      const float4* s = (const float4*)(adjRow + k0 + 64);
      g0 = s[0]; g1 = s[1]; g2 = s[2]; g3 = s[3];
    }

    const _Float16* A = ldsA[p];
    #pragma unroll
    for (int ks = 0; ks < 64; ks += 32) {
      v16h a0 = load_a_frag(&A[(m16     ) * SA2 + ks], hk);
      v16h a1 = load_a_frag(&A[(m16 + 16) * SA2 + ks], hk);
      v16h a2 = load_a_frag(&A[(m16 + 32) * SA2 + ks], hk);
      v16h a3 = load_a_frag(&A[(m16 + 48) * SA2 + ks], hk);
      #pragma unroll
      for (int i = 0; i < 2; ++i) {
        const int cb = wave * 2 + i;
        v16h bf = load_b_frag(xTB + (size_t)(cb * 16 + m16) * 4096
                              + k0 + ks + kh16);
        acc[0 * 2 + i] = wmma_f16(a0, bf, acc[0 * 2 + i]);
        acc[1 * 2 + i] = wmma_f16(a1, bf, acc[1 * 2 + i]);
        acc[2 * 2 + i] = wmma_f16(a2, bf, acc[2 * 2 + i]);
        acc[3 * 2 + i] = wmma_f16(a3, bf, acc[3 * 2 + i]);
      }
    }

    if (more) {
      // direct 2D shared indexing -> stays in LDS addr space (ds_store)
      _Float16* d = &ldsA[p ^ 1][stOff];
      d[0]  = (f16)g0.x; d[1]  = (f16)g0.y; d[2]  = (f16)g0.z; d[3]  = (f16)g0.w;
      d[4]  = (f16)g1.x; d[5]  = (f16)g1.y; d[6]  = (f16)g1.z; d[7]  = (f16)g1.w;
      d[8]  = (f16)g2.x; d[9]  = (f16)g2.y; d[10] = (f16)g2.z; d[11] = (f16)g2.w;
      d[12] = (f16)g3.x; d[13] = (f16)g3.y; d[14] = (f16)g3.z; d[15] = (f16)g3.w;
    }
    __syncthreads();
  }

  // epilogue: h = (1+eps)*x + agg, store f16
  const float scale  = 1.0f + epsp[0];
  const int   rowOff = (lane >> 4) * 8;
  #pragma unroll
  for (int rb = 0; rb < 4; ++rb) {
    #pragma unroll
    for (int i = 0; i < 2; ++i) {
      const int col = (wave * 2 + i) * 16 + m16;
      #pragma unroll
      for (int r = 0; r < 8; ++r) {
        const int row = n0 + rb * 16 + rowOff + r;
        float v = acc[rb * 2 + i][r] + scale * xB[(size_t)row * 256 + col];
        hhB[(size_t)row * 256 + col] = (f16)v;
      }
    }
  }
}

// ---------------------------------------------------------------------------
// Stage 2: out = relu(h @ W1 + b1) @ W2 + b2, 64 rows/WG (halves L2 weight
// traffic). h-tile and hidden-tile alias one LDS buffer (disjoint phases).
// ---------------------------------------------------------------------------
#define SH  272  // h tile row stride in halves (256+16)
#define SH2 528  // hidden tile row stride in halves (512+16)

__global__ __launch_bounds__(256)
void gin_mlp_kernel(const f16* __restrict__ hh, const f16* __restrict__ w1t,
                    const float* __restrict__ b1, const f16* __restrict__ w2t,
                    const float* __restrict__ b2, float* __restrict__ out) {
  const int row0 = blockIdx.x * 64;  // over 8*4096 rows
  const int tid  = threadIdx.x;
  const int wave = tid >> 5;
  const int lane = tid & 31;
  const int m16  = lane & 15;
  const int hk   = (lane >> 4) * 8;
  const int kh16 = (lane >> 4) * 16;
  const int rowOff = (lane >> 4) * 8;

  __shared__ _Float16 smem[64 * SH2];  // 66KB; aliased: [64][SH] then [64][SH2]
  _Float16* ldsH  = smem;
  _Float16* ldsH2 = smem;

  // cooperative load h tile [64][256] f16 (128B/thread, coalesced)
  {
    const int r = tid >> 2;
    const int c = (tid & 3) * 64;
    const v8h* src = (const v8h*)(hh + (size_t)(row0 + r) * 256 + c);
    v8h* dst = (v8h*)&smem[r * SH + c];
    #pragma unroll
    for (int j = 0; j < 8; ++j) dst[j] = src[j];
  }
  __syncthreads();

  const v8f vz = {};
  v8f acc1[16];  // [rb 0..3][cb 0..3]
  #pragma unroll
  for (int i = 0; i < 16; ++i) acc1[i] = vz;

  for (int k0 = 0; k0 < 256; k0 += 32) {
    v16h a0 = load_a_frag(&ldsH[(m16     ) * SH + k0], hk);
    v16h a1 = load_a_frag(&ldsH[(m16 + 16) * SH + k0], hk);
    v16h a2 = load_a_frag(&ldsH[(m16 + 32) * SH + k0], hk);
    v16h a3 = load_a_frag(&ldsH[(m16 + 48) * SH + k0], hk);
    #pragma unroll
    for (int i = 0; i < 4; ++i) {
      const int n = (wave * 4 + i) * 16 + m16;
      v16h bf = load_b_frag(w1t + (size_t)n * 256 + k0 + kh16);
      acc1[0 * 4 + i] = wmma_f16(a0, bf, acc1[0 * 4 + i]);
      acc1[1 * 4 + i] = wmma_f16(a1, bf, acc1[1 * 4 + i]);
      acc1[2 * 4 + i] = wmma_f16(a2, bf, acc1[2 * 4 + i]);
      acc1[3 * 4 + i] = wmma_f16(a3, bf, acc1[3 * 4 + i]);
    }
  }
  __syncthreads();  // all waves done reading ldsH before hidden tile overwrite

  // bias + ReLU -> hidden tile in (aliased) LDS, f16
  #pragma unroll
  for (int rb = 0; rb < 4; ++rb) {
    #pragma unroll
    for (int i = 0; i < 4; ++i) {
      const int col = (wave * 4 + i) * 16 + m16;
      const float bias = b1[col];
      #pragma unroll
      for (int r = 0; r < 8; ++r) {
        const int row = rb * 16 + rowOff + r;
        float v = acc1[rb * 4 + i][r] + bias;
        ldsH2[row * SH2 + col] = (f16)fmaxf(v, 0.0f);
      }
    }
  }
  __syncthreads();

  v8f acc2[8];  // [rb 0..3][cb 0..1]
  #pragma unroll
  for (int i = 0; i < 8; ++i) acc2[i] = vz;

  for (int k0 = 0; k0 < 512; k0 += 32) {
    v16h a0 = load_a_frag(&ldsH2[(m16     ) * SH2 + k0], hk);
    v16h a1 = load_a_frag(&ldsH2[(m16 + 16) * SH2 + k0], hk);
    v16h a2 = load_a_frag(&ldsH2[(m16 + 32) * SH2 + k0], hk);
    v16h a3 = load_a_frag(&ldsH2[(m16 + 48) * SH2 + k0], hk);
    #pragma unroll
    for (int i = 0; i < 2; ++i) {
      const int n = (wave * 2 + i) * 16 + m16;
      v16h bf = load_b_frag(w2t + (size_t)n * 512 + k0 + kh16);
      acc2[0 * 2 + i] = wmma_f16(a0, bf, acc2[0 * 2 + i]);
      acc2[1 * 2 + i] = wmma_f16(a1, bf, acc2[1 * 2 + i]);
      acc2[2 * 2 + i] = wmma_f16(a2, bf, acc2[2 * 2 + i]);
      acc2[3 * 2 + i] = wmma_f16(a3, bf, acc2[3 * 2 + i]);
    }
  }

  #pragma unroll
  for (int rb = 0; rb < 4; ++rb) {
    #pragma unroll
    for (int i = 0; i < 2; ++i) {
      const int col = (wave * 2 + i) * 16 + m16;
      const float bias = b2[col];
      #pragma unroll
      for (int r = 0; r < 8; ++r) {
        const int row = row0 + rb * 16 + rowOff + r;
        out[(size_t)row * 256 + col] = acc2[rb * 2 + i][r] + bias;
      }
    }
  }
}

// ---------------------------------------------------------------------------
// Launch. Inputs: x, adj, eps, W1, b1, W2, b2 (all f32). Output f32.
// Workspace: [0,16MB) xT f16 | [16MB,32MB) h f16 | +256KB W1t | +256KB W2t
// ---------------------------------------------------------------------------
extern "C" void kernel_launch(void* const* d_in, const int* in_sizes, int n_in,
                              void* d_out, int out_size, void* d_ws, size_t ws_size,
                              hipStream_t stream) {
  const float* x   = (const float*)d_in[0];
  const float* adj = (const float*)d_in[1];
  const float* eps = (const float*)d_in[2];
  const float* W1  = (const float*)d_in[3];
  const float* b1  = (const float*)d_in[4];
  const float* W2  = (const float*)d_in[5];
  const float* b2  = (const float*)d_in[6];
  float* out = (float*)d_out;

  char* ws = (char*)d_ws;
  f16* xT  = (f16*)(ws);
  f16* hh  = (f16*)(ws + (size_t)16 * 1024 * 1024);
  f16* w1t = (f16*)(ws + (size_t)32 * 1024 * 1024);
  f16* w2t = (f16*)(ws + (size_t)32 * 1024 * 1024 + 256 * 1024);

  transpose_x_kernel<<<dim3(128, 8, 8), dim3(32, 8), 0, stream>>>(x, xT);
  transpose_w_kernel<<<(256 * 512 + 255) / 256, 256, 0, stream>>>(W1, w1t, 256, 512);
  transpose_w_kernel<<<(512 * 256 + 255) / 256, 256, 0, stream>>>(W2, w2t, 512, 256);

  gin_bmm_kernel<<<dim3(64, 8), 256, 0, stream>>>(adj, xT, x, eps, hh);
  gin_mlp_kernel<<<dim3(512), 256, 0, stream>>>(hh, w1t, b1, w2t, b2, out);
}
